// GAT_64390149702047
// MI455X (gfx1250) — matrix operand order repaired and checked by reading, not compiled, not run on hardware
//
#include <hip/hip_runtime.h>
#include <math.h>

typedef __attribute__((ext_vector_type(2)))  float     v2f;
typedef __attribute__((ext_vector_type(8)))  float     v8f;
typedef __attribute__((ext_vector_type(16))) _Float16  v16h;

#if defined(__has_builtin)
#if __has_builtin(__builtin_amdgcn_wmma_f32_16x16x4_f32)
#define GAT_USE_F32_WMMA 1
#else
#define GAT_USE_F32_WMMA 0
#endif
#else
#define GAT_USE_F32_WMMA 0
#endif

#define GAT_LRELU(v) ((v) > 0.f ? (v) : 0.2f * (v))

__device__ __forceinline__ void gat_atomicMaxF(float* addr, float v) {
  // classic float-max idiom: signed int compare for >=0, unsigned min for <0
  if (v >= 0.f) atomicMax((int*)addr, __float_as_int(v));
  else          atomicMin((unsigned int*)addr, __float_as_uint(v));
}

// ---------------------------------------------------------------------------
// WMMA GEMM: Out[n,NC] = X[n,KD] @ W[KD,NC] (+bias) (+relu)
// grid.x = n/16 (n always a multiple of 16 here), block = (NC/16) waves.
// X tile and W^T staged in LDS; one wave per 16x16 output tile; fp32 WMMA
// (16x16x4) keeps reference precision. W transposed in LDS so both A and B
// operands are contiguous K-pairs -> 8-byte-aligned ds_load_b64.
// ---------------------------------------------------------------------------
template <int KD, int NC>
__global__ void gat_gemm_wmma(const float* __restrict__ X, const float* __restrict__ W,
                              const float* __restrict__ bias, float* __restrict__ Out,
                              int relu) {
  constexpr int LDK = KD + 4;               // padded K stride (even)
  __shared__ float xs[16 * LDK];            // 16 rows x KD
  __shared__ float wshT[NC * LDK];          // NC cols x KD (transposed W)
  const int tid  = threadIdx.x;
  const int lane = tid & 31;
  const int wave = tid >> 5;
  const int row0 = blockIdx.x * 16;
  constexpr int NTHR = NC * 2;              // (NC/16) waves

  for (int i = tid; i < KD * NC; i += NTHR) {
    int k = i / NC, col = i - k * NC;       // coalesced global read
    wshT[col * LDK + k] = W[i];             // transposed LDS write
  }
  for (int i = tid; i < 16 * KD; i += NTHR) {
    int r = i / KD, c = i - r * KD;
    xs[r * LDK + c] = X[(size_t)(row0 + r) * KD + c];
  }
  __syncthreads();

  const int m       = lane & 15;            // M (for A) / N (for B,C)
  const int colbase = wave * 16 + m;        // output column
  v8f acc = {};

#if GAT_USE_F32_WMMA
  const int koff = (lane < 16) ? 0 : 2;     // f32 A/B lane->K mapping (16x16x4)
  const float* aBase = xs   + m * LDK       + koff;   // 8B aligned
  const float* bBase = wshT + colbase * LDK + koff;   // 8B aligned
#pragma unroll
  for (int k = 0; k < KD; k += 4) {
    v2f a = *(const v2f*)(aBase + k);
    v2f b = *(const v2f*)(bBase + k);
    acc = __builtin_amdgcn_wmma_f32_16x16x4_f32(false, a, false, b, (short)0, acc,
                                                false, false);
  }
#else
  // f16 fallback (KD is always a multiple of 32 here)
  const float* aBase = xs   + m * LDK;
  const float* bBase = wshT + colbase * LDK;
#pragma unroll
  for (int k = 0; k < KD; k += 32) {
    v16h a, b;
#pragma unroll
    for (int j = 0; j < 16; ++j) {
      int vg = j >> 1, p = j & 1;
      int ka = (vg < 4 ? vg * 2 + p : 16 + (vg - 4) * 2 + p) + ((lane < 16) ? 0 : 8) + k;
      a[j] = (_Float16)aBase[ka];
      int kb = (j < 8 ? j : j + 8) + ((lane < 16) ? 0 : 8) + k;
      b[j] = (_Float16)bBase[kb];
    }
    acc = __builtin_amdgcn_wmma_f32_16x16x32_f16(false, a, false, b, (short)0, acc,
                                                 false, false);
  }
#endif

  const int mb = (lane < 16) ? 0 : 8;
  const float bv = bias ? bias[colbase] : 0.f;
#pragma unroll
  for (int r = 0; r < 8; ++r) {
    float v = acc[r] + bv;
    if (relu) v = v > 0.f ? v : 0.f;
    Out[(size_t)(row0 + mb + r) * NC + colbase] = v;
  }
}

static void gat_launch_gemm(hipStream_t st, const float* X, const float* W,
                            const float* bias, float* Out, int n, int Kdim, int Ncols,
                            int relu) {
  if (Kdim == 128 && Ncols == 96)
    gat_gemm_wmma<128, 96><<<n / 16, 192, 0, st>>>(X, W, bias, Out, relu);
  else if (Kdim == 32 && Ncols == 96)
    gat_gemm_wmma<32, 96><<<n / 16, 192, 0, st>>>(X, W, bias, Out, relu);
  else if (Kdim == 96 && Ncols == 32)
    gat_gemm_wmma<96, 32><<<n / 16, 64, 0, st>>>(X, W, bias, Out, relu);
  else if (Kdim == 64 && Ncols == 32)
    gat_gemm_wmma<64, 32><<<n / 16, 64, 0, st>>>(X, W, bias, Out, relu);
}

// ---------------------------------------------------------------------------
// GAT attention kernels
// ---------------------------------------------------------------------------
__global__ void gat_attn_logits(const float* __restrict__ h, const float* __restrict__ a_s,
                                const float* __restrict__ a_d,
                                float* __restrict__ als, float* __restrict__ ald, int n3) {
  int i = blockIdx.x * blockDim.x + threadIdx.x;
  if (i >= n3) return;
  int node = i / 3, hh = i - node * 3;
  const float* hp  = h + (size_t)node * 96 + hh * 32;
  const float* asv = a_s + hh * 32;
  const float* adv = a_d + hh * 32;
  float ss = 0.f, sd = 0.f;
  for (int c = 0; c < 32; ++c) { float v = hp[c]; ss += v * asv[c]; sd += v * adv[c]; }
  als[i] = ss; ald[i] = sd;
}

__global__ void gat_init_mx(const float* __restrict__ als, const float* __restrict__ ald,
                            float* __restrict__ mx, int n3) {
  int i = blockIdx.x * blockDim.x + threadIdx.x;
  if (i >= n3) return;
  float v = als[i] + ald[i];               // self-loop logit
  mx[i] = GAT_LRELU(v);
}

__global__ void gat_edge_max(const int* __restrict__ src, const int* __restrict__ dst,
                             const int* __restrict__ mask,
                             const float* __restrict__ als, const float* __restrict__ ald,
                             float* __restrict__ mx, int E) {
  int e = blockIdx.x * blockDim.x + threadIdx.x;
  if (e >= E) return;
  if (mask && !mask[e]) return;
  int s = src[e], d = dst[e];
  for (int hh = 0; hh < 3; ++hh) {
    float v = als[s * 3 + hh] + ald[d * 3 + hh];
    gat_atomicMaxF(&mx[d * 3 + hh], GAT_LRELU(v));
  }
}

__global__ void gat_den_self(const float* __restrict__ als, const float* __restrict__ ald,
                             const float* __restrict__ mx, float* __restrict__ den, int n3) {
  int i = blockIdx.x * blockDim.x + threadIdx.x;
  if (i >= n3) return;
  float v = als[i] + ald[i];
  den[i] = expf(GAT_LRELU(v) - mx[i]);     // self-loop contribution
}

__global__ void gat_edge_den(const int* __restrict__ src, const int* __restrict__ dst,
                             const int* __restrict__ mask,
                             const float* __restrict__ als, const float* __restrict__ ald,
                             const float* __restrict__ mx, float* __restrict__ den, int E) {
  int e = blockIdx.x * blockDim.x + threadIdx.x;
  if (e >= E) return;
  if (mask && !mask[e]) return;
  int s = src[e], d = dst[e];
  for (int hh = 0; hh < 3; ++hh) {
    float v = als[s * 3 + hh] + ald[d * 3 + hh];
    float ex = expf(GAT_LRELU(v) - mx[d * 3 + hh]);
    atomicAdd(&den[d * 3 + hh], ex);
  }
}

// gout[n,96] = b + alpha_self * h  (thread per (node,feature))
__global__ void gat_self_agg(const float* __restrict__ h,
                             const float* __restrict__ als, const float* __restrict__ ald,
                             const float* __restrict__ mx, const float* __restrict__ den,
                             const float* __restrict__ bvec, float* __restrict__ gout,
                             int total) {
  int idx = blockIdx.x * blockDim.x + threadIdx.x;
  if (idx >= total) return;
  int node = idx / 96, r = idx - node * 96, hh = r >> 5;
  int i = node * 3 + hh;
  float v = als[i] + ald[i];
  float alpha = expf(GAT_LRELU(v) - mx[i]) / (den[i] + 1e-16f);
  gout[idx] = bvec[r] + alpha * h[idx];
}

// gout[d] += alpha * h[s]  (thread per (edge,head), 32 feature adds each)
__global__ void gat_edge_agg(const int* __restrict__ src, const int* __restrict__ dst,
                             const int* __restrict__ mask,
                             const float* __restrict__ als, const float* __restrict__ ald,
                             const float* __restrict__ mx, const float* __restrict__ den,
                             const float* __restrict__ h, float* __restrict__ gout, int E) {
  int t = blockIdx.x * blockDim.x + threadIdx.x;
  if (t >= 3 * E) return;
  int e = t % E, hh = t / E;               // consecutive threads -> consecutive edges
  if (mask && !mask[e]) return;
  int s = src[e], d = dst[e];
  const float* hp = h + (size_t)s * 96 + hh * 32;
  __builtin_prefetch(hp, 0, 1);            // global_prefetch_b8
  int i = d * 3 + hh;
  float v = als[s * 3 + hh] + ald[i];
  float alpha = expf(GAT_LRELU(v) - mx[i]) / (den[i] + 1e-16f);
  float* op = gout + (size_t)d * 96 + hh * 32;
  for (int c = 0; c < 32; ++c) atomicAdd(&op[c], alpha * hp[c]);
}

// ---------------------------------------------------------------------------
// TopK pooling: one block per graph, rank-by-counting (tie-break = lower idx)
// ---------------------------------------------------------------------------
__global__ void gat_topk_pool(const float* __restrict__ x, const float* __restrict__ pw,
                              int n_per, int K, float* __restrict__ xnew,
                              int* __restrict__ newid) {
  __shared__ float sc[512];
  int b = blockIdx.x, t = threadIdx.x;
  float wn = 0.f;
  for (int c = 0; c < 32; ++c) wn += pw[c] * pw[c];
  float rn = 1.0f / sqrtf(wn);
  int node = b * n_per + t;
  float s = 0.f;
  if (t < n_per) {
    float dot = 0.f;
    for (int c = 0; c < 32; ++c) dot += x[(size_t)node * 32 + c] * pw[c];
    s = tanhf(dot * rn);
    sc[t] = s;
  }
  __syncthreads();
  if (t >= n_per) return;
  int rank = 0;
  for (int j = 0; j < n_per; ++j) {
    float o = sc[j];
    rank += (o > s) || (o == s && j < t);
  }
  if (rank < K) {
    int np = b * K + rank;
    newid[node] = np;
    for (int c = 0; c < 32; ++c) xnew[(size_t)np * 32 + c] = x[(size_t)node * 32 + c] * s;
  } else {
    newid[node] = -1;
  }
}

__global__ void gat_remap_edges(const int* __restrict__ srcI, const int* __restrict__ dstI,
                                const int* __restrict__ maskI, const int* __restrict__ newid,
                                int* __restrict__ srcO, int* __restrict__ dstO,
                                int* __restrict__ maskO, int E) {
  int e = blockIdx.x * blockDim.x + threadIdx.x;
  if (e >= E) return;
  int valid = maskI ? maskI[e] : 1;
  int ns = 0, nd = 0;
  if (valid) {
    ns = newid[srcI[e]];
    nd = newid[dstI[e]];
    valid = (ns >= 0) && (nd >= 0);
  }
  srcO[e]  = valid ? ns : 0;
  dstO[e]  = valid ? nd : 0;
  maskO[e] = valid;
}

// oacc[b, 0:32] += max_k, oacc[b, 32:64] += mean_k
__global__ void gat_readout(const float* __restrict__ xnew, int K, float* __restrict__ oacc) {
  int t = blockIdx.x * blockDim.x + threadIdx.x;
  if (t >= 128 * 32) return;
  int b = t >> 5, c = t & 31;
  float mv = -1e30f, sum = 0.f;
  for (int k = 0; k < K; ++k) {
    float v = xnew[(size_t)(b * K + k) * 32 + c];
    mv = fmaxf(mv, v);
    sum += v;
  }
  oacc[b * 64 + c]      += mv;
  oacc[b * 64 + 32 + c] += sum / (float)K;
}

__global__ void gat_zero(float* __restrict__ p, int n) {
  int i = blockIdx.x * blockDim.x + threadIdx.x;
  if (i < n) p[i] = 0.f;
}

__global__ void gat_final(const float* __restrict__ hid, const float* __restrict__ outW,
                          const float* __restrict__ outb, float* __restrict__ out) {
  int b = blockIdx.x * blockDim.x + threadIdx.x;
  if (b >= 128) return;
  float s = 0.f;
  for (int j = 0; j < 32; ++j) s += hid[b * 32 + j] * outW[j];
  out[b] = s + outb[0];
}

// ---------------------------------------------------------------------------
// Host-side stage helper (all launches on `st`)
// ---------------------------------------------------------------------------
static void gat_run_stage(hipStream_t st, const float* x, int Kin, const float* W,
                          const float* a_s, const float* a_d, const float* bvec,
                          const float* headW, const float* headb,
                          const int* src, const int* dst, const int* mask, int E, int n,
                          float* hbuf, float* gout, float* als, float* ald,
                          float* mx, float* den, float* hproj) {
  const int TH = 256;
  gat_launch_gemm(st, x, W, nullptr, hbuf, n, Kin, 96, 0);
  int n3 = n * 3;
  gat_attn_logits<<<(n3 + TH - 1) / TH, TH, 0, st>>>(hbuf, a_s, a_d, als, ald, n3);
  gat_init_mx<<<(n3 + TH - 1) / TH, TH, 0, st>>>(als, ald, mx, n3);
  gat_edge_max<<<(E + TH - 1) / TH, TH, 0, st>>>(src, dst, mask, als, ald, mx, E);
  gat_den_self<<<(n3 + TH - 1) / TH, TH, 0, st>>>(als, ald, mx, den, n3);
  gat_edge_den<<<(E + TH - 1) / TH, TH, 0, st>>>(src, dst, mask, als, ald, mx, den, E);
  gat_self_agg<<<(n * 96 + TH - 1) / TH, TH, 0, st>>>(hbuf, als, ald, mx, den, bvec, gout,
                                                      n * 96);
  gat_edge_agg<<<(3 * E + TH - 1) / TH, TH, 0, st>>>(src, dst, mask, als, ald, mx, den,
                                                     hbuf, gout, E);
  gat_launch_gemm(st, gout, headW, headb, hproj, n, 96, 32, 0);
}

extern "C" void kernel_launch(void* const* d_in, const int* in_sizes, int n_in,
                              void* d_out, int out_size, void* d_ws, size_t ws_size,
                              hipStream_t stream) {
  const float* x     = (const float*)d_in[0];
  const int*   ei    = (const int*)d_in[1];
  const float* W_in  = (const float*)d_in[3];
  const float* as0   = (const float*)d_in[4];
  const float* ad0   = (const float*)d_in[5];
  const float* b0    = (const float*)d_in[6];
  const float* W1    = (const float*)d_in[7];
  const float* as1   = (const float*)d_in[8];
  const float* ad1   = (const float*)d_in[9];
  const float* b1    = (const float*)d_in[10];
  const float* W2    = (const float*)d_in[11];
  const float* as2   = (const float*)d_in[12];
  const float* ad2   = (const float*)d_in[13];
  const float* b2    = (const float*)d_in[14];
  const float* headW = (const float*)d_in[15];
  const float* headb = (const float*)d_in[16];
  const float* p1    = (const float*)d_in[17];
  const float* p2    = (const float*)d_in[18];
  const float* p3    = (const float*)d_in[19];
  const float* lin1W = (const float*)d_in[20];
  const float* lin1b = (const float*)d_in[21];
  const float* outW  = (const float*)d_in[22];
  const float* outb  = (const float*)d_in[23];

  const int N  = in_sizes[0] / 128;     // 65536
  const int E  = in_sizes[1] / 2;       // 1048576
  const int B  = 128;
  const int N0 = 512, K1 = 410, K2 = 205, K3 = 41;
  const int n1 = B * K1, n2 = B * K2;   // 52480, 26240 (multiples of 16)

  // -------- workspace carve (deterministic) --------
  char* w = (char*)d_ws;
  float* hbuf  = (float*)w; w += (size_t)N * 96 * 4;
  float* gout  = (float*)w; w += (size_t)N * 96 * 4;
  float* hproj = (float*)w; w += (size_t)N * 32 * 4;
  float* xnext = (float*)w; w += (size_t)N * 32 * 4;
  float* als   = (float*)w; w += (size_t)N * 3 * 4;
  float* ald   = (float*)w; w += (size_t)N * 3 * 4;
  float* mxb   = (float*)w; w += (size_t)N * 3 * 4;
  float* den   = (float*)w; w += (size_t)N * 3 * 4;
  int*   newid = (int*)w;   w += (size_t)N * 4;
  int*   srcA  = (int*)w;   w += (size_t)E * 4;
  int*   dstA  = (int*)w;   w += (size_t)E * 4;
  int*   mskA  = (int*)w;   w += (size_t)E * 4;
  int*   srcB  = (int*)w;   w += (size_t)E * 4;
  int*   dstB  = (int*)w;   w += (size_t)E * 4;
  int*   mskB  = (int*)w;   w += (size_t)E * 4;
  float* oacc  = (float*)w; w += (size_t)B * 64 * 4;
  float* hid   = (float*)w; w += (size_t)B * 32 * 4;
  (void)ws_size; (void)n_in; (void)out_size;

  const int* src0 = ei;
  const int* dst0 = ei + E;
  const int TH = 256;

  gat_zero<<<(B * 64 + TH - 1) / TH, TH, 0, stream>>>(oacc, B * 64);

  // ---- stage 1: GAT(FEAT=128) + head, pool K1, readout ----
  gat_run_stage(stream, x, 128, W_in, as0, ad0, b0, headW, headb,
                src0, dst0, nullptr, E, N, hbuf, gout, als, ald, mxb, den, hproj);
  gat_topk_pool<<<B, 512, 0, stream>>>(hproj, p1, N0, K1, xnext, newid);
  gat_remap_edges<<<(E + TH - 1) / TH, TH, 0, stream>>>(src0, dst0, nullptr, newid,
                                                        srcA, dstA, mskA, E);
  gat_readout<<<(B * 32 + TH - 1) / TH, TH, 0, stream>>>(xnext, K1, oacc);

  // ---- stage 2: GAT(C=32) + head, pool K2, readout ----
  gat_run_stage(stream, xnext, 32, W1, as1, ad1, b1, headW, headb,
                srcA, dstA, mskA, E, n1, hbuf, gout, als, ald, mxb, den, hproj);
  gat_topk_pool<<<B, 512, 0, stream>>>(hproj, p2, K1, K2, xnext, newid);
  gat_remap_edges<<<(E + TH - 1) / TH, TH, 0, stream>>>(srcA, dstA, mskA, newid,
                                                        srcB, dstB, mskB, E);
  gat_readout<<<(B * 32 + TH - 1) / TH, TH, 0, stream>>>(xnext, K2, oacc);

  // ---- stage 3: GAT(C=32) + head, pool K3, readout ----
  gat_run_stage(stream, xnext, 32, W2, as2, ad2, b2, headW, headb,
                srcB, dstB, mskB, E, n2, hbuf, gout, als, ald, mxb, den, hproj);
  gat_topk_pool<<<B, 512, 0, stream>>>(hproj, p3, K2, K3, xnext, newid);
  gat_readout<<<(B * 32 + TH - 1) / TH, TH, 0, stream>>>(xnext, K3, oacc);

  // ---- MLP head: relu(o @ lin1_W + lin1_b) @ out_W + out_b ----
  gat_launch_gemm(stream, oacc, lin1W, lin1b, hid, B, 64, 32, 1);
  gat_final<<<1, 128, 0, stream>>>(hid, outW, outb, (float*)d_out);
}
